// ReaderUnit_24739011625869
// MI455X (gfx1250) — matrix operand order repaired and checked by reading, not compiled, not run on hardware
//
#include <hip/hip_runtime.h>
#include <hip/hip_bf16.h>
#include <math.h>

// ---------------------------------------------------------------------------
// Types for CDNA5 WMMA (wave32): D = A(16x32 bf16) x B(32x16 bf16) + C(16x16 f32)
// ---------------------------------------------------------------------------
typedef __attribute__((ext_vector_type(16))) __bf16 v16bf;
typedef __attribute__((ext_vector_type(8)))  float  v8f;

#define DEVINL static __device__ __forceinline__

DEVINL unsigned short f2bf(float f) {
  union { float f; unsigned u; } x; x.f = f;
  unsigned u = x.u;
  unsigned r = u + 0x7FFFu + ((u >> 16) & 1u);       // round-to-nearest-even
  if ((u & 0x7F800000u) == 0x7F800000u) r = u;       // inf/nan passthrough
  return (unsigned short)(r >> 16);
}

// ---------------------------------------------------------------------------
// f32 -> bf16 conversion kernel (weights / inputs)
// ---------------------------------------------------------------------------
__global__ __launch_bounds__(256) void k_f2bf(const float* __restrict__ x,
                                              unsigned short* __restrict__ y, int n) {
  int i = blockIdx.x * blockDim.x + threadIdx.x;
  if (i < n) y[i] = f2bf(x[i]);
}

// ---------------------------------------------------------------------------
// GEMM NN:  C[M,N] = act( scale * (A[M,K] @ B[K,N]) + bias[N] )
//   A,B bf16 row-major (u16), C fp32, optional bf16 mirror Cbf.
//   act: 0 none, 1 PReLU (alpha read from device pointer).
//   Requirements: K % 32 == 0, N % 32 == 0. M, N(per 64-tile) guarded.
//   Block: 256 thr = 8 wave32; tile 64(M) x 64(N); each wave 16x32 (2 accs).
//   Double-buffered LDS; B staged TRANSPOSED (Bs[n][k]) so both A and B
//   fragments are two aligned ds_load_b128 each.
// ---------------------------------------------------------------------------
#define LDSW 40  // padded row stride in ushorts: 80B, 16B aligned, bank-skewed

__global__ __launch_bounds__(256) void k_gemm_nn(
    const unsigned short* __restrict__ A, int lda,
    const unsigned short* __restrict__ B, int ldb,
    float* __restrict__ C, int ldc,
    unsigned short* __restrict__ Cbf, int ldcbf,
    const float* __restrict__ bias,
    int M, int N, int K,
    int act, const float* __restrict__ alphaPtr, float scale) {
  __shared__ unsigned short As[2][64 * LDSW];      // [row][k]
  __shared__ unsigned short Bs[2][64 * LDSW];      // [n][k]  (transposed)
  const int t    = threadIdx.x;
  const int lane = t & 31;
  const int wid  = t >> 5;
  const int wm   = wid & 3;                        // 16-row tile
  const int wn   = wid >> 2;                       // 32-col slab
  const int rowBase = blockIdx.y * 64;
  const int colBase = blockIdx.x * 64;

  v8f acc0 = {0.f, 0.f, 0.f, 0.f, 0.f, 0.f, 0.f, 0.f};
  v8f acc1 = acc0;

  // A staging: 64 rows x 32 k; thread -> one uint4 (8 ushorts)
  const int aRow = t >> 2;
  const int aSeg = (t & 3) * 8;
  // B staging: 32 k x 64 n; thread loads 8 consecutive n at one k, then
  // transposes into Bs[n][k] with scalar stores.
  const int bK   = t >> 3;
  const int bSeg = (t & 7) * 8;

  const int lr    = lane & 15;
  const int kbase = (lane >> 4) * 16;              // A-frag K half (contiguous)
  const int khalf = (lane >> 4) * 16;              // B-frag K half

  const int nSteps = K / 32;

  // ---- stage tile 0 into buffer 0 ----
  {
    int gr = rowBase + aRow;
    uint4 va = make_uint4(0u, 0u, 0u, 0u);
    if (gr < M) va = *(const uint4*)(A + (size_t)gr * lda + aSeg);
    *(uint4*)(&As[0][aRow * LDSW + aSeg]) = va;

    uint4 vb = make_uint4(0u, 0u, 0u, 0u);
    int gcol = colBase + bSeg;
    if (gcol < N) vb = *(const uint4*)(B + (size_t)bK * ldb + gcol);
    const unsigned short* pv = (const unsigned short*)&vb;
#pragma unroll
    for (int i = 0; i < 8; ++i) Bs[0][(bSeg + i) * LDSW + bK] = pv[i];
  }

  for (int s = 0; s < nSteps; ++s) {
    __syncthreads();
    const int cur = s & 1;
    // ---- stage tile s+1 into the other buffer (overlaps WMMA below) ----
    if (s + 1 < nSteps) {
      const int nxt = cur ^ 1;
      const int k0  = (s + 1) * 32;
      int gr = rowBase + aRow;
      uint4 va = make_uint4(0u, 0u, 0u, 0u);
      if (gr < M) {
        va = *(const uint4*)(A + (size_t)gr * lda + k0 + aSeg);
        if (s + 2 < nSteps)
          __builtin_prefetch(A + (size_t)gr * lda + k0 + 32 + aSeg, 0, 1);
      }
      *(uint4*)(&As[nxt][aRow * LDSW + aSeg]) = va;

      uint4 vb = make_uint4(0u, 0u, 0u, 0u);
      int gcol = colBase + bSeg;
      if (gcol < N) {
        vb = *(const uint4*)(B + (size_t)(k0 + bK) * ldb + gcol);
        if (s + 2 < nSteps)
          __builtin_prefetch(B + (size_t)(k0 + 32 + bK) * ldb + gcol, 0, 1);
      }
      const unsigned short* pv = (const unsigned short*)&vb;
#pragma unroll
      for (int i = 0; i < 8; ++i) Bs[nxt][(bSeg + i) * LDSW + bK] = pv[i];
    }
    // ---- fragments (all ds_load_b128) + 2 WMMA ----
    union { unsigned short u[16]; v16bf v; } fa, fb0, fb1;
    {
      const unsigned short* ap = &As[cur][(wm * 16 + lr) * LDSW];
      *(uint4*)&fa.u[0] = *(const uint4*)(ap + kbase);
      *(uint4*)&fa.u[8] = *(const uint4*)(ap + kbase + 8);
    }
    {
      const unsigned short* bp0 = &Bs[cur][(wn * 32 + lr) * LDSW + khalf];
      *(uint4*)&fb0.u[0] = *(const uint4*)(bp0);
      *(uint4*)&fb0.u[8] = *(const uint4*)(bp0 + 8);
      const unsigned short* bp1 = &Bs[cur][(wn * 32 + 16 + lr) * LDSW + khalf];
      *(uint4*)&fb1.u[0] = *(const uint4*)(bp1);
      *(uint4*)&fb1.u[8] = *(const uint4*)(bp1 + 8);
    }
    acc0 = __builtin_amdgcn_wmma_f32_16x16x32_bf16(false, fa.v, false, fb0.v,
                                                   (short)0, acc0, false, false);
    acc1 = __builtin_amdgcn_wmma_f32_16x16x32_bf16(false, fa.v, false, fb1.v,
                                                   (short)0, acc1, false, false);
  }

  // ---- epilogue ----
  const int hi = lane >> 4;
  float alpha = (act == 1 && alphaPtr) ? alphaPtr[0] : 0.f;
#pragma unroll
  for (int half = 0; half < 2; ++half) {
    const v8f* accp = half ? &acc1 : &acc0;
    int col = colBase + wn * 32 + half * 16 + lr;
    if (col < N) {
      float bcol = bias ? bias[col] : 0.f;
#pragma unroll
      for (int i = 0; i < 8; ++i) {
        int row = rowBase + wm * 16 + hi * 8 + i;
        if (row < M) {
          float v = (*accp)[i] * scale + bcol;
          if (act == 1) v = (v >= 0.f) ? v : alpha * v;
          C[(size_t)row * ldc + col] = v;
          if (Cbf) Cbf[(size_t)row * ldcbf + col] = f2bf(v);
        }
      }
    }
  }
}

// ---------------------------------------------------------------------------
// GEMM NT:  C[M,N] = scale * (A[M,K] @ B[N,K]^T)     (attention logits)
//   B tile is naturally [n][k]; staging and fragments fully vectorized.
// ---------------------------------------------------------------------------
__global__ __launch_bounds__(256) void k_gemm_nt(
    const unsigned short* __restrict__ A, int lda,
    const unsigned short* __restrict__ B, int ldb,
    float* __restrict__ C, int ldc,
    int M, int N, int K, float scale) {
  __shared__ unsigned short As[2][64 * LDSW];
  __shared__ unsigned short Bs[2][64 * LDSW];      // [n][k]
  const int t    = threadIdx.x;
  const int lane = t & 31;
  const int wid  = t >> 5;
  const int wm   = wid & 3;
  const int wn   = wid >> 2;
  const int rowBase = blockIdx.y * 64;
  const int colBase = blockIdx.x * 64;

  v8f acc0 = {0.f, 0.f, 0.f, 0.f, 0.f, 0.f, 0.f, 0.f};
  v8f acc1 = acc0;

  const int aRow = t >> 2;
  const int aSeg = (t & 3) * 8;
  const int bN   = t >> 2;                         // 0..63 (n index)
  const int bSeg = (t & 3) * 8;                    // k segment

  const int lr    = lane & 15;
  const int kbase = (lane >> 4) * 16;
  const int khalf = (lane >> 4) * 16;

  const int nSteps = K / 32;

  {
    int gr = rowBase + aRow;
    uint4 va = make_uint4(0u, 0u, 0u, 0u);
    if (gr < M) va = *(const uint4*)(A + (size_t)gr * lda + aSeg);
    *(uint4*)(&As[0][aRow * LDSW + aSeg]) = va;

    int gn = colBase + bN;
    uint4 vb = make_uint4(0u, 0u, 0u, 0u);
    if (gn < N) vb = *(const uint4*)(B + (size_t)gn * ldb + bSeg);
    *(uint4*)(&Bs[0][bN * LDSW + bSeg]) = vb;
  }

  for (int s = 0; s < nSteps; ++s) {
    __syncthreads();
    const int cur = s & 1;
    if (s + 1 < nSteps) {
      const int nxt = cur ^ 1;
      const int k0  = (s + 1) * 32;
      int gr = rowBase + aRow;
      uint4 va = make_uint4(0u, 0u, 0u, 0u);
      if (gr < M) va = *(const uint4*)(A + (size_t)gr * lda + k0 + aSeg);
      *(uint4*)(&As[nxt][aRow * LDSW + aSeg]) = va;

      int gn = colBase + bN;
      uint4 vb = make_uint4(0u, 0u, 0u, 0u);
      if (gn < N) vb = *(const uint4*)(B + (size_t)gn * ldb + k0 + bSeg);
      *(uint4*)(&Bs[nxt][bN * LDSW + bSeg]) = vb;
    }
    union { unsigned short u[16]; v16bf v; } fa, fb0, fb1;
    {
      const unsigned short* ap = &As[cur][(wm * 16 + lr) * LDSW];
      *(uint4*)&fa.u[0] = *(const uint4*)(ap + kbase);
      *(uint4*)&fa.u[8] = *(const uint4*)(ap + kbase + 8);
    }
    {
      const unsigned short* bp0 = &Bs[cur][(wn * 32 + lr) * LDSW + khalf];
      *(uint4*)&fb0.u[0] = *(const uint4*)(bp0);
      *(uint4*)&fb0.u[8] = *(const uint4*)(bp0 + 8);
      const unsigned short* bp1 = &Bs[cur][(wn * 32 + 16 + lr) * LDSW + khalf];
      *(uint4*)&fb1.u[0] = *(const uint4*)(bp1);
      *(uint4*)&fb1.u[8] = *(const uint4*)(bp1 + 8);
    }
    acc0 = __builtin_amdgcn_wmma_f32_16x16x32_bf16(false, fa.v, false, fb0.v,
                                                   (short)0, acc0, false, false);
    acc1 = __builtin_amdgcn_wmma_f32_16x16x32_bf16(false, fa.v, false, fb1.v,
                                                   (short)0, acc1, false, false);
  }

  const int hi = lane >> 4;
#pragma unroll
  for (int half = 0; half < 2; ++half) {
    const v8f* accp = half ? &acc1 : &acc0;
    int col = colBase + wn * 32 + half * 16 + lr;
    if (col < N) {
#pragma unroll
      for (int i = 0; i < 8; ++i) {
        int row = rowBase + wm * 16 + hi * 8 + i;
        if (row < M) C[(size_t)row * ldc + col] = (*accp)[i] * scale;
      }
    }
  }
}

// ---------------------------------------------------------------------------
// Row softmax, in place; optional bf16 mirror. One block per row.
// ---------------------------------------------------------------------------
__global__ __launch_bounds__(256) void k_softmax(float* __restrict__ X,
                                                 unsigned short* __restrict__ Xbf,
                                                 int cols) {
  float* x = X + (size_t)blockIdx.x * cols;
  unsigned short* xb = Xbf ? Xbf + (size_t)blockIdx.x * cols : nullptr;
  __shared__ float red[256];
  int t = threadIdx.x;
  float m = -3.4e38f;
  for (int c = t; c < cols; c += 256) m = fmaxf(m, x[c]);
  red[t] = m; __syncthreads();
  for (int s = 128; s > 0; s >>= 1) { if (t < s) red[t] = fmaxf(red[t], red[t + s]); __syncthreads(); }
  m = red[0]; __syncthreads();
  float sum = 0.f;
  for (int c = t; c < cols; c += 256) { float e = __expf(x[c] - m); x[c] = e; sum += e; }
  red[t] = sum; __syncthreads();
  for (int s = 128; s > 0; s >>= 1) { if (t < s) red[t] += red[t + s]; __syncthreads(); }
  float inv = 1.f / red[0];
  for (int c = t; c < cols; c += 256) {
    float v = x[c] * inv;
    x[c] = v;
    if (xb) xb[c] = f2bf(v);
  }
}

// ---------------------------------------------------------------------------
// LayerNorm family. One block (256 thr) per row.
//   mode 0: out = LN(x)*g+be + res          (FFN residual)
//   mode 1: out = PReLU(LN(x*x2)*g+be)      (pair context)
//   mode 2: out = PReLU(LN(x)*g+be)         (fo head)
// ---------------------------------------------------------------------------
__global__ __launch_bounds__(256) void k_ln(
    const float* __restrict__ X, const float* __restrict__ X2,
    const float* __restrict__ res,
    const float* __restrict__ g, const float* __restrict__ be,
    const float* __restrict__ alphaPtr,
    float* __restrict__ outF, unsigned short* __restrict__ outBf,
    int cols, int mode) {
  size_t row = blockIdx.x;
  const float* x  = X + row * cols;
  const float* x2 = X2 ? X2 + row * cols : nullptr;
  __shared__ float red[256];
  __shared__ float stats[2];
  int t = threadIdx.x;
  float s = 0.f, sq = 0.f;
  for (int c = t; c < cols; c += 256) {
    float v = x[c]; if (x2) v *= x2[c];
    s += v; sq += v * v;
  }
  red[t] = s; __syncthreads();
  for (int k = 128; k; k >>= 1) { if (t < k) red[t] += red[t + k]; __syncthreads(); }
  if (t == 0) stats[0] = red[0] / (float)cols;
  __syncthreads();
  red[t] = sq; __syncthreads();
  for (int k = 128; k; k >>= 1) { if (t < k) red[t] += red[t + k]; __syncthreads(); }
  if (t == 0) {
    float mean = stats[0];
    float var  = fmaxf(red[0] / (float)cols - mean * mean, 0.f);
    stats[1] = rsqrtf(var + 1e-5f);
  }
  __syncthreads();
  float mean = stats[0], inv = stats[1];
  float alpha = alphaPtr ? alphaPtr[0] : 0.f;
  for (int c = t; c < cols; c += 256) {
    float v = x[c]; if (x2) v *= x2[c];
    float y = (v - mean) * inv * g[c] + be[c];
    if (mode >= 1) y = (y >= 0.f) ? y : alpha * y;
    if (mode == 0 && res) y += res[row * cols + c];
    if (outF)  outF[row * cols + c] = y;
    if (outBf) outBf[row * cols + c] = f2bf(y);
  }
}

// ---------------------------------------------------------------------------
// rb[(n*256+m)] = concat(kb[m][0:1024], embed[|tf[n]-tb[m]|][0:128])
// writes fp32 + bf16 mirrors. One block per row (8192 rows).
// ---------------------------------------------------------------------------
__global__ __launch_bounds__(256) void k_build_rb(
    const float* __restrict__ kb,       // [256,1024] batch slice (fp32)
    const float* __restrict__ embed,    // [65,128]
    const int* __restrict__ tb,         // bank times [256]
    const int* __restrict__ tf,         // feat times [32]
    float* __restrict__ rb, unsigned short* __restrict__ rbbf) {
  int row = blockIdx.x;
  int n = row >> 8, m = row & 255;
  int d = tf[n] - tb[m]; d = d < 0 ? -d : d;
  const float* krow = kb + (size_t)m * 1024;
  const float* erow = embed + (size_t)d * 128;
  float* dst = rb + (size_t)row * 1152;
  unsigned short* dstb = rbbf + (size_t)row * 1152;
  for (int c = threadIdx.x; c < 1024; c += 256) { float v = krow[c]; dst[c] = v; dstb[c] = f2bf(v); }
  for (int c = threadIdx.x; c < 128;  c += 256) { float v = erow[c]; dst[1024 + c] = v; dstb[1024 + c] = f2bf(v); }
}

// ---------------------------------------------------------------------------
// logits[n*256+m] = scale * dot(fk[n,:512], kbp[n*256+m,:512]); one wave/dot
// ---------------------------------------------------------------------------
__global__ __launch_bounds__(256) void k_qk_dot(
    const float* __restrict__ fk,       // [32,512] batch slice
    const float* __restrict__ kbp,      // [8192,512]
    float* __restrict__ logits, float scale) {
  int idx  = blockIdx.x * 8 + (threadIdx.x >> 5);
  int lane = threadIdx.x & 31;
  int n = idx >> 8;
  const float* a = fk  + (size_t)n * 512;
  const float* b = kbp + (size_t)idx * 512;
  float s = 0.f;
  for (int c = lane; c < 512; c += 32) s += a[c] * b[c];
  for (int off = 16; off; off >>= 1) s += __shfl_xor(s, off, 32);
  if (lane == 0) logits[idx] = s * scale;
}

// ---------------------------------------------------------------------------
// fo[n,l] = sum_m coeff[n,m] * fbp[n*256+m, l]
// ---------------------------------------------------------------------------
__global__ __launch_bounds__(256) void k_attn_out(
    const float* __restrict__ coeff,    // [32,256]
    const float* __restrict__ fbp,      // [8192,512]
    float* __restrict__ fo) {           // [32,512]
  int n = blockIdx.y;
  int l = blockIdx.x * 256 + threadIdx.x;
  const float* c = coeff + n * 256;
  const float* f = fbp + (size_t)n * 256 * 512 + l;
  float s = 0.f;
  for (int m = 0; m < 256; ++m) s += c[m] * f[(size_t)m * 512];
  fo[n * 512 + l] = s;
}

// ---------------------------------------------------------------------------
// y = (zero? 0 : y) + agg[p] * x   (pair aggregation)
// ---------------------------------------------------------------------------
__global__ __launch_bounds__(256) void k_axpy_agg(
    float* __restrict__ y, const float* __restrict__ x,
    const float* __restrict__ agg, int p, int n, int zero) {
  int i = blockIdx.x * 256 + threadIdx.x;
  if (i < n) {
    float w = agg[p];
    y[i] = (zero ? 0.f : y[i]) + w * x[i];
  }
}

// ---------------------------------------------------------------------------
// out[base+n] = feat[base+n] + fod[n] + pc[n]      (rows of 1024)
// ---------------------------------------------------------------------------
__global__ __launch_bounds__(256) void k_final_add(
    const float* __restrict__ feat, const float* __restrict__ fod,
    const float* __restrict__ pc, float* __restrict__ out, int base) {
  int n = blockIdx.x;
  size_t row = (size_t)(base + n);
  for (int c = threadIdx.x; c < 1024; c += 256)
    out[row * 1024 + c] = feat[row * 1024 + c] + fod[n * 1024 + c] + pc[n * 1024 + c];
}

// ===========================================================================
// Host orchestration
// ===========================================================================
// Flattened input indices (setup_inputs dict insertion order, recursively):
#define IN_FEAT   0
#define IN_KB     1
#define IN_TS     2
#define IN_BIDX   3
#define IN_P1W    4
#define IN_P1B    5
#define IN_P2W    6
#define IN_P2B    7
#define IN_P3W    8
#define IN_P3B    9
#define IN_FFN_G  10
#define IN_FFN_B  11
#define IN_FFN_A  12
#define IN_FFN_W  13
#define IN_FFN_WB 14
#define IN_AGG    15
#define IN_EMBED  16
#define IN_CODEC0 17   // +0 w1, +1 b1, +2 a, +3 w2, +4 b2, +5 g, +6 be
#define IN_CODEC1 24
#define IN_PAIR0  31   // +0 c1w,+1 c1b,+2 c2w,+3 c2b,+4 c3w,+5 c3b,
                       // +6 d1w,+7 d1b,+8 d2w,+9 d2b,+10 d3w,+11 d3b,
                       // +12 g,+13 be,+14 a,+15 w,+16 wb
#define IN_PAIR1  48

static inline void launch_gemm_nn(hipStream_t st,
    const unsigned short* A, int lda, const unsigned short* B, int ldb,
    float* C, int ldc, unsigned short* Cbf, int ldcbf,
    const float* bias, int M, int N, int K,
    int act, const float* alphaPtr, float scale) {
  dim3 grid((N + 63) / 64, (M + 63) / 64);
  k_gemm_nn<<<grid, 256, 0, st>>>(A, lda, B, ldb, C, ldc, Cbf, ldcbf,
                                  bias, M, N, K, act, alphaPtr, scale);
}

static inline void launch_gemm_nt(hipStream_t st,
    const unsigned short* A, int lda, const unsigned short* B, int ldb,
    float* C, int ldc, int M, int N, int K, float scale) {
  dim3 grid((N + 63) / 64, (M + 63) / 64);
  k_gemm_nt<<<grid, 256, 0, st>>>(A, lda, B, ldb, C, ldc, M, N, K, scale);
}

static inline void launch_f2bf(hipStream_t st, const float* s, unsigned short* d, int n) {
  k_f2bf<<<(n + 255) / 256, 256, 0, st>>>(s, d, n);
}

extern "C" void kernel_launch(void* const* d_in, const int* in_sizes, int n_in,
                              void* d_out, int out_size, void* d_ws, size_t ws_size,
                              hipStream_t stream) {
  (void)in_sizes; (void)n_in; (void)out_size; (void)ws_size;
  const int Dd = 1024, Ll = 512, Bb = 8, Nn = 32, Mm = 256, MD = 1152, Hh = 288;
  const float SCL = 1.0f / sqrtf((float)Ll);

  const float* feat   = (const float*)d_in[IN_FEAT];
  const float* keyb   = (const float*)d_in[IN_KB];
  const int*   ts     = (const int*)d_in[IN_TS];
  const float* embed  = (const float*)d_in[IN_EMBED];
  const float* agg    = (const float*)d_in[IN_AGG];
  float* out = (float*)d_out;

  // ---- workspace bump allocator (256B aligned) ----
  char* wsb = (char*)d_ws;
  size_t off = 0;
  auto alloc = [&](size_t bytes) -> void* {
    void* p = wsb + off;
    off = (off + bytes + 255) & ~(size_t)255;
    return p;
  };

  unsigned short* featbf = (unsigned short*)alloc((size_t)Bb * Nn * Dd * 2);
  unsigned short* kbbf   = (unsigned short*)alloc((size_t)Bb * Mm * Dd * 2);
  float*          fkeyF  = (float*)alloc((size_t)Bb * Nn * Ll * 4);

  // bf16 weight copies
  unsigned short* wp1  = (unsigned short*)alloc((size_t)Dd * Ll * 2);
  unsigned short* wp2  = (unsigned short*)alloc((size_t)MD * Ll * 2);
  unsigned short* wp3  = (unsigned short*)alloc((size_t)MD * Ll * 2);
  unsigned short* wffn = (unsigned short*)alloc((size_t)Ll * Dd * 2);
  unsigned short* wc1[2], *wc2[2];
  for (int j = 0; j < 2; ++j) {
    wc1[j] = (unsigned short*)alloc((size_t)MD * Hh * 2);
    wc2[j] = (unsigned short*)alloc((size_t)Hh * MD * 2);
  }
  unsigned short* pwm[2][6];  // c1w,c2w,c3w,d1w,d2w,d3w
  unsigned short* pwo[2];     // output w
  for (int p = 0; p < 2; ++p) {
    for (int q = 0; q < 6; ++q) pwm[p][q] = (unsigned short*)alloc((size_t)Dd * Ll * 2);
    pwo[p] = (unsigned short*)alloc((size_t)Ll * Dd * 2);
  }

  // per-batch scratch (reused across the 8 sequential batches)
  float*          pairCtx = (float*)alloc((size_t)Nn * Dd * 4);
  float*          projF   = (float*)alloc((size_t)Mm * Ll * 4);
  unsigned short* rBf[2]  = {(unsigned short*)alloc((size_t)Nn * Ll * 2),
                             (unsigned short*)alloc((size_t)Nn * Ll * 2)};
  unsigned short* kBf[2]  = {(unsigned short*)alloc((size_t)Mm * Ll * 2),
                             (unsigned short*)alloc((size_t)Mm * Ll * 2)};
  unsigned short* vBf[2]  = {(unsigned short*)alloc((size_t)Mm * Ll * 2),
                             (unsigned short*)alloc((size_t)Mm * Ll * 2)};
  float*          attnF   = (float*)alloc((size_t)Nn * Mm * 4);
  unsigned short* probsBf = (unsigned short*)alloc((size_t)Nn * Mm * 2);
  float*          f1F     = (float*)alloc((size_t)Nn * Ll * 4);
  float*          f2F     = (float*)alloc((size_t)Nn * Ll * 4);
  unsigned short* cBf     = (unsigned short*)alloc((size_t)Nn * Ll * 2);
  float*          coutF   = (float*)alloc((size_t)Nn * Dd * 4);

  float*          rbF  = (float*)alloc((size_t)Nn * Mm * MD * 4);
  unsigned short* rbBf = (unsigned short*)alloc((size_t)Nn * Mm * MD * 2);
  float*          hF   = (float*)alloc((size_t)Nn * Mm * Hh * 4);
  unsigned short* hBf  = (unsigned short*)alloc((size_t)Nn * Mm * Hh * 2);
  float*          h2F  = (float*)alloc((size_t)Nn * Mm * MD * 4);
  float*          kbpF = (float*)alloc((size_t)Nn * Mm * Ll * 4);
  float*          fbpF = (float*)alloc((size_t)Nn * Mm * Ll * 4);
  float*          lg2F = (float*)alloc((size_t)Nn * Mm * 4);
  float*          foF  = (float*)alloc((size_t)Nn * Ll * 4);
  unsigned short* foBf = (unsigned short*)alloc((size_t)Nn * Ll * 2);
  float*          fodF = (float*)alloc((size_t)Nn * Dd * 4);

  // ---- weight / input conversions (fp32 -> bf16) ----
  launch_f2bf(stream, feat, featbf, Bb * Nn * Dd);
  launch_f2bf(stream, keyb, kbbf, Bb * Mm * Dd);
  launch_f2bf(stream, (const float*)d_in[IN_P1W], wp1, Dd * Ll);
  launch_f2bf(stream, (const float*)d_in[IN_P2W], wp2, MD * Ll);
  launch_f2bf(stream, (const float*)d_in[IN_P3W], wp3, MD * Ll);
  launch_f2bf(stream, (const float*)d_in[IN_FFN_W], wffn, Ll * Dd);
  for (int j = 0; j < 2; ++j) {
    int cb = (j == 0) ? IN_CODEC0 : IN_CODEC1;
    launch_f2bf(stream, (const float*)d_in[cb + 0], wc1[j], MD * Hh);
    launch_f2bf(stream, (const float*)d_in[cb + 3], wc2[j], Hh * MD);
  }
  for (int p = 0; p < 2; ++p) {
    int pb = (p == 0) ? IN_PAIR0 : IN_PAIR1;
    for (int q = 0; q < 6; ++q)
      launch_f2bf(stream, (const float*)d_in[pb + 2 * q], pwm[p][q], Dd * Ll);
    launch_f2bf(stream, (const float*)d_in[pb + 15], pwo[p], Ll * Dd);
  }

  // ---- fkey = feat @ p1w + p1b  [256,512] ----
  launch_gemm_nn(stream, featbf, Dd, wp1, Ll, fkeyF, Ll, nullptr, 0,
                 (const float*)d_in[IN_P1B], Bb * Nn, Ll, Dd, 0, nullptr, 1.f);

  for (int b = 0; b < Bb; ++b) {
    const unsigned short* fb_bf = featbf + (size_t)b * Nn * Dd;  // [32,1024]
    const unsigned short* kb_bf = kbbf + (size_t)b * Mm * Dd;    // [256,1024]

    // ============ pair attention blocks ============
    for (int p = 0; p < 2; ++p) {
      int pb = (p == 0) ? IN_PAIR0 : IN_PAIR1;
      // r1 = fb@c1w+c1b ; r2 = fb@d1w+d1b
      launch_gemm_nn(stream, fb_bf, Dd, pwm[p][0], Ll, projF, Ll, rBf[0], Ll,
                     (const float*)d_in[pb + 1], Nn, Ll, Dd, 0, nullptr, 1.f);
      launch_gemm_nn(stream, fb_bf, Dd, pwm[p][3], Ll, projF, Ll, rBf[1], Ll,
                     (const float*)d_in[pb + 7], Nn, Ll, Dd, 0, nullptr, 1.f);
      // k1 = kb@c2w+c2b ; v1 = kb@c3w+c3b ; k2 = kb@d2w+d2b ; v2 = kb@d3w+d3b
      launch_gemm_nn(stream, kb_bf, Dd, pwm[p][1], Ll, projF, Ll, kBf[0], Ll,
                     (const float*)d_in[pb + 3], Mm, Ll, Dd, 0, nullptr, 1.f);
      launch_gemm_nn(stream, kb_bf, Dd, pwm[p][2], Ll, projF, Ll, vBf[0], Ll,
                     (const float*)d_in[pb + 5], Mm, Ll, Dd, 0, nullptr, 1.f);
      launch_gemm_nn(stream, kb_bf, Dd, pwm[p][4], Ll, projF, Ll, kBf[1], Ll,
                     (const float*)d_in[pb + 9], Mm, Ll, Dd, 0, nullptr, 1.f);
      launch_gemm_nn(stream, kb_bf, Dd, pwm[p][5], Ll, projF, Ll, vBf[1], Ll,
                     (const float*)d_in[pb + 11], Mm, Ll, Dd, 0, nullptr, 1.f);

      // f1 = softmax(r1 k1^T * s) v1 ; f2 likewise
      for (int h = 0; h < 2; ++h) {
        launch_gemm_nt(stream, rBf[h], Ll, kBf[h], Ll, attnF, Mm, Nn, Mm, Ll, SCL);
        k_softmax<<<Nn, 256, 0, stream>>>(attnF, probsBf, Mm);
        launch_gemm_nn(stream, probsBf, Mm, vBf[h], Ll,
                       (h == 0 ? f1F : f2F), Ll, nullptr, 0,
                       nullptr, Nn, Ll, Mm, 0, nullptr, 1.f);
      }
      // c = PReLU(LN(f1*f2)) -> bf16
      k_ln<<<Nn, 256, 0, stream>>>(f1F, f2F, nullptr,
                                   (const float*)d_in[pb + 12],
                                   (const float*)d_in[pb + 13],
                                   (const float*)d_in[pb + 14],
                                   nullptr, cBf, Ll, 1);
      // cout = c @ w + wb
      launch_gemm_nn(stream, cBf, Ll, pwo[p], Dd, coutF, Dd, nullptr, 0,
                     (const float*)d_in[pb + 16], Nn, Dd, Ll, 0, nullptr, 1.f);
      // pair_ctx (+)= agg[p] * cout
      k_axpy_agg<<<(Nn * Dd + 255) / 256, 256, 0, stream>>>(
          pairCtx, coutF, agg, p, Nn * Dd, p == 0);
    }

    // ============ codec / temporal stage ============
    const int* tb = ts + (size_t)b * (Mm + Nn);
    const int* tf = tb + Mm;
    k_build_rb<<<Nn * Mm, 256, 0, stream>>>(keyb + (size_t)b * Mm * Dd,
                                            embed, tb, tf, rbF, rbBf);
    for (int j = 0; j < 2; ++j) {
      int cb = (j == 0) ? IN_CODEC0 : IN_CODEC1;
      // h = PReLU(rb @ w1 + b1, a)  -> fp32 + bf16 mirror
      launch_gemm_nn(stream, rbBf, MD, wc1[j], Hh, hF, Hh, hBf, Hh,
                     (const float*)d_in[cb + 1], Nn * Mm, Hh, MD,
                     1, (const float*)d_in[cb + 2], 1.f);
      // h2 = h @ w2 + b2
      launch_gemm_nn(stream, hBf, Hh, wc2[j], MD, h2F, MD, nullptr, 0,
                     (const float*)d_in[cb + 4], Nn * Mm, MD, Hh, 0, nullptr, 1.f);
      // rb = LN(h2)*g+be + rb  (in place), refresh bf16 mirror
      k_ln<<<Nn * Mm, 256, 0, stream>>>(h2F, nullptr, rbF,
                                        (const float*)d_in[cb + 5],
                                        (const float*)d_in[cb + 6],
                                        nullptr, rbF, rbBf, MD, 0);
    }
    // kbp = rb @ p2w + p2b ; fbp = rb @ p3w + p3b
    launch_gemm_nn(stream, rbBf, MD, wp2, Ll, kbpF, Ll, nullptr, 0,
                   (const float*)d_in[IN_P2B], Nn * Mm, Ll, MD, 0, nullptr, 1.f);
    launch_gemm_nn(stream, rbBf, MD, wp3, Ll, fbpF, Ll, nullptr, 0,
                   (const float*)d_in[IN_P3B], Nn * Mm, Ll, MD, 0, nullptr, 1.f);

    // coeff = softmax(fk . kbp * s) ; fo = coeff @ fbp
    k_qk_dot<<<(Nn * Mm) / 8, 256, 0, stream>>>(fkeyF + (size_t)b * Nn * Ll,
                                                kbpF, lg2F, SCL);
    k_softmax<<<Nn, 256, 0, stream>>>(lg2F, nullptr, Mm);
    {
      dim3 g(Ll / 256, Nn);
      k_attn_out<<<g, 256, 0, stream>>>(lg2F, fbpF, foF);
    }
    // fo = PReLU(LN(fo)*g+b, a) -> bf16
    k_ln<<<Nn, 256, 0, stream>>>(foF, nullptr, nullptr,
                                 (const float*)d_in[IN_FFN_G],
                                 (const float*)d_in[IN_FFN_B],
                                 (const float*)d_in[IN_FFN_A],
                                 nullptr, foBf, Ll, 2);
    // fod = fo @ ffn_w + ffn_wb
    launch_gemm_nn(stream, foBf, Ll, wffn, Dd, fodF, Dd, nullptr, 0,
                   (const float*)d_in[IN_FFN_WB], Nn, Dd, Ll, 0, nullptr, 1.f);
    // out rows = feat + fod + pair_ctx
    k_final_add<<<Nn, 256, 0, stream>>>(feat, fodF, pairCtx, out, b * Nn);
  }
}
